// Block_89988154785920
// MI455X (gfx1250) — compile-verified, matrix operand order checked by reading, not run on hardware
//
#include <hip/hip_runtime.h>
#include <hip/hip_bf16.h>

typedef __bf16 bf16;
typedef __attribute__((ext_vector_type(16))) __bf16 v16bf;
typedef __attribute__((ext_vector_type(8)))  float  v8f;

union ABFrag { v16bf v; uint4 q[2]; };

#define B_DIM 2
#define T_DIM 2048
#define C_DIM 1024
#define NT (B_DIM * T_DIM)
#define HQ 16
#define HKV 8
#define HD 64
#define WINDOW 512
#define HIDDEN 4096
#define SOFT_CAP 50.0f

static __device__ inline v8f vzero8() { v8f v; for (int i = 0; i < 8; ++i) v[i] = 0.f; return v; }

static __device__ inline v8f wmma_bf16(v16bf a, v16bf b, v8f c) {
  return __builtin_amdgcn_wmma_f32_16x16x32_bf16(false, a, false, b, (short)0, c, false, false);
}

// gfx1250 async global->LDS copy (16B per lane), tracked by ASYNCcnt
static __device__ inline void async_g2l_b128(unsigned int lds_addr, const void* gptr) {
  asm volatile("global_load_async_to_lds_b128 %0, %1, off"
               :: "v"(lds_addr), "v"(gptr) : "memory");
}
static __device__ inline unsigned int lds_off(const void* p) {
  return (unsigned int)(unsigned long long)p;   // low 32 bits = LDS offset
}
static __device__ inline void wait_asynccnt0() {
  asm volatile("s_wait_asynccnt 0x0" ::: "memory");
}
static __device__ inline void wait_dscnt0() {
  asm volatile("s_wait_dscnt 0x0" ::: "memory");
}

// ---------------- block reduction helper ----------------
static __device__ inline float block_reduce_sum_256(float v) {
  __shared__ float red[8];
  int lane = threadIdx.x & 31, w = threadIdx.x >> 5;
  for (int o = 16; o; o >>= 1) v += __shfl_down(v, o, 32);
  if (lane == 0) red[w] = v;
  __syncthreads();
  float r = (threadIdx.x < 8) ? red[threadIdx.x] : 0.f;
  if (w == 0) for (int o = 4; o; o >>= 1) r += __shfl_down(r, o, 32);
  if (threadIdx.x == 0) red[0] = r;
  __syncthreads();
  r = red[0];
  __syncthreads();
  return r;
}

// ---------------- f32 -> bf16 convert ----------------
__global__ void cvt_bf16_kernel(const float* __restrict__ in, bf16* __restrict__ out, int n) {
  int i = blockIdx.x * blockDim.x + threadIdx.x;
  if (i < n) out[i] = (bf16)in[i];
}

// ---------------- RMSNorm -> bf16 ----------------
__global__ __launch_bounds__(256) void rmsnorm_kernel(const float* __restrict__ X,
                                                      const float* __restrict__ scale,
                                                      bf16* __restrict__ O) {
  int row = blockIdx.x;
  const float* x = X + (size_t)row * C_DIM;
  float ss = 0.f;
  for (int c = threadIdx.x; c < C_DIM; c += 256) { float v = x[c]; ss += v * v; }
  float tot = block_reduce_sum_256(ss);
  float rstd = rsqrtf(tot * (1.0f / C_DIM) + 1e-6f);
  for (int c = threadIdx.x; c < C_DIM; c += 256)
    O[(size_t)row * C_DIM + c] = (bf16)(x[c] * rstd * (1.f + scale[c]));
}

// ---------------- generic bf16 GEMM: out = A[MxK] @ W[KxN] (+res) ----------------
#define GT_M 64
#define GT_N 64
#define GT_K 32
#define GLDS 40   // halves per row (80B: 16B aligned, bank-staggered)

template <int M, int N, int K>
__global__ __launch_bounds__(128) void gemm_bf16_kernel(
    const bf16* __restrict__ A, const bf16* __restrict__ W,
    float* __restrict__ Cf, bf16* __restrict__ Cb, const float* __restrict__ Res) {
  __shared__ __align__(16) bf16 sA[GT_M * GLDS];
  __shared__ __align__(16) bf16 sB[GT_N * GLDS];   // transposed [n][k]

  const int tid = threadIdx.x;
  const int wave = tid >> 5, lane = tid & 31;
  const int row0 = blockIdx.y * GT_M, col0 = blockIdx.x * GT_N;
  const int wm = (wave >> 1) * 32, wn = (wave & 1) * 32;
  const int mr = lane & 15;
  const int kb = (lane >> 4) << 3;

  // per-thread fixed staging slots (A tile: 2 x 16B chunks)
  const int ar0 = tid >> 2, ac = tid & 3;
  const int ar1 = ar0 + 32;
  const unsigned int sa0 = lds_off(&sA[ar0 * GLDS + ac * 8]);
  const unsigned int sa1 = lds_off(&sA[ar1 * GLDS + ac * 8]);
  const bf16* ap0 = A + (size_t)(row0 + ar0) * K + ac * 8;
  const bf16* ap1 = A + (size_t)(row0 + ar1) * K + ac * 8;

  // B transpose-staging: nl invariant, kl steps by 2 per iteration (immediate offsets)
  const int nl = tid & 63, klb = tid >> 6;
  const bf16* wp = W + (size_t)klb * N + col0 + nl;
  bf16* sbp = &sB[nl * GLDS + klb];

  v8f acc[2][2];
  for (int i = 0; i < 2; ++i) for (int j = 0; j < 2; ++j) acc[i][j] = vzero8();

  for (int k0 = 0; k0 < K; k0 += GT_K) {
    // --- A tile: async global->LDS, no VGPR round-trip ---
    async_g2l_b128(sa0, ap0);
    async_g2l_b128(sa1, ap1);
    ap0 += GT_K; ap1 += GT_K;

    // --- B tile: constant-offset loads (all independent), then LDS store phase ---
    bf16 btmp[16];
#pragma unroll
    for (int it = 0; it < 16; ++it) btmp[it] = wp[(size_t)it * 2 * N];
#pragma unroll
    for (int it = 0; it < 16; ++it) sbp[it * 2] = btmp[it];
    wp += (size_t)GT_K * N;

    if (k0 + GT_K < K) {   // prefetch next A tile rows into cache
      __builtin_prefetch(ap0, 0, 0);
      __builtin_prefetch(ap1, 0, 0);
    }

    wait_asynccnt0();
    __syncthreads();

    ABFrag af[2], bfr[2];
#pragma unroll
    for (int i = 0; i < 2; ++i) {
      int m = wm + i * 16 + mr;
      af[i].q[0] = *(const uint4*)(&sA[m * GLDS + kb]);
      af[i].q[1] = *(const uint4*)(&sA[m * GLDS + kb + 16]);
    }
#pragma unroll
    for (int j = 0; j < 2; ++j) {
      int n = wn + j * 16 + mr;
      bfr[j].q[0] = *(const uint4*)(&sB[n * GLDS + kb]);
      bfr[j].q[1] = *(const uint4*)(&sB[n * GLDS + kb + 16]);
    }
#pragma unroll
    for (int i = 0; i < 2; ++i)
#pragma unroll
      for (int j = 0; j < 2; ++j)
        acc[i][j] = wmma_bf16(af[i].v, bfr[j].v, acc[i][j]);
    __syncthreads();
  }

  const int mb = (lane >> 4) * 8;
  for (int i = 0; i < 2; ++i) for (int j = 0; j < 2; ++j) {
    int gm0 = row0 + wm + i * 16 + mb;
    int gn  = col0 + wn + j * 16 + mr;
    size_t idx = (size_t)gm0 * N + gn;
#pragma unroll
    for (int r = 0; r < 8; ++r) {
      float v = acc[i][j][r];
      size_t id2 = idx + (size_t)r * N;
      if (Res) v += Res[id2];
      if (Cf) Cf[id2] = v;
      if (Cb) Cb[id2] = (bf16)v;
    }
  }
}

// ---------------- fused gate/up GEMM: H = silu(A@Wg) * (A@Wu), bf16 ----------------
template <int M, int N, int K>
__global__ __launch_bounds__(128) void mlp_gateup_kernel(
    const bf16* __restrict__ A, const bf16* __restrict__ Wg, const bf16* __restrict__ Wu,
    bf16* __restrict__ H) {
  __shared__ __align__(16) bf16 sA[GT_M * GLDS];
  __shared__ __align__(16) bf16 sG[GT_N * GLDS];
  __shared__ __align__(16) bf16 sU[GT_N * GLDS];

  const int tid = threadIdx.x;
  const int wave = tid >> 5, lane = tid & 31;
  const int row0 = blockIdx.y * GT_M, col0 = blockIdx.x * GT_N;
  const int wm = (wave >> 1) * 32, wn = (wave & 1) * 32;
  const int mr = lane & 15;
  const int kb = (lane >> 4) << 3;

  const int ar0 = tid >> 2, ac = tid & 3;
  const int ar1 = ar0 + 32;
  const unsigned int sa0 = lds_off(&sA[ar0 * GLDS + ac * 8]);
  const unsigned int sa1 = lds_off(&sA[ar1 * GLDS + ac * 8]);
  const bf16* ap0 = A + (size_t)(row0 + ar0) * K + ac * 8;
  const bf16* ap1 = A + (size_t)(row0 + ar1) * K + ac * 8;

  const int nl = tid & 63, klb = tid >> 6;
  const bf16* gp = Wg + (size_t)klb * N + col0 + nl;
  const bf16* up = Wu + (size_t)klb * N + col0 + nl;
  bf16* sgp = &sG[nl * GLDS + klb];
  bf16* sup = &sU[nl * GLDS + klb];

  v8f ag[2][2], au[2][2];
  for (int i = 0; i < 2; ++i) for (int j = 0; j < 2; ++j) { ag[i][j] = vzero8(); au[i][j] = vzero8(); }

  for (int k0 = 0; k0 < K; k0 += GT_K) {
    async_g2l_b128(sa0, ap0);
    async_g2l_b128(sa1, ap1);
    ap0 += GT_K; ap1 += GT_K;

    bf16 gtmp[16], utmp[16];
#pragma unroll
    for (int it = 0; it < 16; ++it) { gtmp[it] = gp[(size_t)it * 2 * N]; utmp[it] = up[(size_t)it * 2 * N]; }
#pragma unroll
    for (int it = 0; it < 16; ++it) { sgp[it * 2] = gtmp[it]; sup[it * 2] = utmp[it]; }
    gp += (size_t)GT_K * N;
    up += (size_t)GT_K * N;

    wait_asynccnt0();
    __syncthreads();

    ABFrag af[2], gf[2], uf[2];
#pragma unroll
    for (int i = 0; i < 2; ++i) {
      int m = wm + i * 16 + mr;
      af[i].q[0] = *(const uint4*)(&sA[m * GLDS + kb]);
      af[i].q[1] = *(const uint4*)(&sA[m * GLDS + kb + 16]);
    }
#pragma unroll
    for (int j = 0; j < 2; ++j) {
      int n = wn + j * 16 + mr;
      gf[j].q[0] = *(const uint4*)(&sG[n * GLDS + kb]);
      gf[j].q[1] = *(const uint4*)(&sG[n * GLDS + kb + 16]);
      uf[j].q[0] = *(const uint4*)(&sU[n * GLDS + kb]);
      uf[j].q[1] = *(const uint4*)(&sU[n * GLDS + kb + 16]);
    }
#pragma unroll
    for (int i = 0; i < 2; ++i)
#pragma unroll
      for (int j = 0; j < 2; ++j) {
        ag[i][j] = wmma_bf16(af[i].v, gf[j].v, ag[i][j]);
        au[i][j] = wmma_bf16(af[i].v, uf[j].v, au[i][j]);
      }
    __syncthreads();
  }

  const int mb = (lane >> 4) * 8;
  for (int i = 0; i < 2; ++i) for (int j = 0; j < 2; ++j) {
    int gm0 = row0 + wm + i * 16 + mb;
    int gn  = col0 + wn + j * 16 + mr;
    size_t idx = (size_t)gm0 * N + gn;
#pragma unroll
    for (int r = 0; r < 8; ++r) {
      float g = ag[i][j][r];
      float sil = g * (1.0f / (1.0f + __expf(-g)));
      H[idx + (size_t)r * N] = (bf16)(sil * au[i][j][r]);
    }
  }
}

// ---------------- RoPE: f32 raw -> bf16 roped ----------------
__global__ void rope_kernel(const float* __restrict__ In, bf16* __restrict__ Out, int nheads) {
  int idx = blockIdx.x * blockDim.x + threadIdx.x;
  int total = NT * nheads * 32;
  if (idx >= total) return;
  int i = idx & 31;
  int h = (idx >> 5) % nheads;
  int bt = idx / (32 * nheads);
  int t = bt & (T_DIM - 1);
  float freq = __expf(-(float)i * (1.0f / 32.0f) * 13.815510557964274f); // ln(1e6)
  float ang = (float)t * freq;
  float s = __sinf(ang), c = __cosf(ang);
  size_t base = (size_t)bt * (nheads * HD) + h * HD + i;
  float x1 = In[base], x2 = In[base + 32];
  Out[base]      = (bf16)(x1 * c - x2 * s);
  Out[base + 32] = (bf16)(x2 * c + x1 * s);
}

// ---------------- sliding-window flash attention, one wave per 16-query tile ----------------
__global__ __launch_bounds__(32) void attn_kernel(
    const bf16* __restrict__ Q, const bf16* __restrict__ Kk,
    const bf16* __restrict__ V, bf16* __restrict__ Ctx) {
  const int lane = threadIdx.x;
  const int qt = blockIdx.x, hq = blockIdx.y, b = blockIdx.z;
  const int kh = hq & 7;
  const int bt0 = b * T_DIM + qt * 16;
  const int mr = lane & 15;
  const int half = lane >> 4;
  const int kb = half << 3;

  __shared__ __align__(16) bf16 sV[32 * 72];   // [d][v], stride 72 halves (144B)
  __shared__ __align__(16) bf16 sP[16 * 40];   // [m][k],  stride 40 halves (80B)

  // Q fragments (persistent over kv loop); contraction dim contiguous in global
  ABFrag qa[2];
  {
    const bf16* qrow = Q + (size_t)(bt0 + mr) * (HQ * HD) + hq * HD;
    qa[0].q[0] = *(const uint4*)(qrow + kb);
    qa[0].q[1] = *(const uint4*)(qrow + kb + 16);
    qa[1].q[0] = *(const uint4*)(qrow + 32 + kb);
    qa[1].q[1] = *(const uint4*)(qrow + 32 + kb + 16);
  }

  v8f O[4];
  for (int t = 0; t < 4; ++t) O[t] = vzero8();
  float Mrow[8], Lrow[8];
  for (int r = 0; r < 8; ++r) { Mrow[r] = -1e30f; Lrow[r] = 0.f; }

  const int iq_lo = qt * 16;
  int jlo = iq_lo - (WINDOW - 1); if (jlo < 0) jlo = 0;
  const int c0 = jlo >> 5;
  const int c1 = (iq_lo + 15) >> 5;

  // running pointers over kv chunks (stride 32 rows)
  const bf16* kp = Kk + (size_t)(b * T_DIM + c0 * 32 + mr) * (HKV * HD) + kh * HD;
  const bf16* vp = V + (size_t)(b * T_DIM + c0 * 32 + (lane >> 3)) * (HKV * HD) + kh * HD + (lane & 7) * 8;
  const unsigned int sv0 = lds_off(&sV[(lane >> 3) * 72 + (lane & 7) * 8]);
  const size_t kchunk = (size_t)32 * (HKV * HD);

  for (int ch = c0; ch <= c1; ++ch) {
    const int j0 = ch * 32;

    // stage V chunk [32 x 64] into LDS via async copies (8 x 16B per lane, stride-4 rows)
#pragma unroll
    for (int it = 0; it < 8; ++it)
      async_g2l_b128(sv0 + it * (4 * 72 * 2), vp + (size_t)it * 4 * (HKV * HD));
    vp += kchunk;

    // scores: 2 sub-tiles of 16 keys, K=64 contraction -> 2 WMMAs each
    v8f Sacc[2];
#pragma unroll
    for (int s = 0; s < 2; ++s) {
      Sacc[s] = vzero8();
      const bf16* krow = kp + (size_t)(s * 16) * (HKV * HD);
      ABFrag kf0, kf1;
      kf0.q[0] = *(const uint4*)(krow + kb);
      kf0.q[1] = *(const uint4*)(krow + kb + 16);
      kf1.q[0] = *(const uint4*)(krow + 32 + kb);
      kf1.q[1] = *(const uint4*)(krow + 32 + kb + 16);
      Sacc[s] = wmma_bf16(qa[0].v, kf0.v, Sacc[s]);
      Sacc[s] = wmma_bf16(qa[1].v, kf1.v, Sacc[s]);
    }
    kp += kchunk;

    // softcap + band mask + online softmax; P -> LDS (bf16, A-fragment friendly layout)
#pragma unroll
    for (int r = 0; r < 8; ++r) {
      int i = iq_lo + r + 8 * half;
      float sv[2];
      float mx = -1e30f;
      for (int s = 0; s < 2; ++s) {
        int j = j0 + s * 16 + mr;
        float x = Sacc[s][r] * 0.125f;                  // 1/sqrt(64)
        x = SOFT_CAP * tanhf(x * (1.0f / SOFT_CAP));
        bool ok = (j <= i) && (i - j < WINDOW);
        x = ok ? x : -1e30f;
        sv[s] = x;
        mx = fmaxf(mx, x);
      }
      for (int o = 8; o; o >>= 1) mx = fmaxf(mx, __shfl_xor(mx, o, 16));
      float Mn = fmaxf(Mrow[r], mx);
      float alpha = __expf(Mrow[r] - Mn);
      float rs = 0.f;
      for (int s = 0; s < 2; ++s) {
        float p = __expf(sv[s] - Mn);
        rs += p;
        sP[(r + 8 * half) * 40 + s * 16 + mr] = (bf16)p;
      }
      for (int o = 8; o; o >>= 1) rs += __shfl_xor(rs, o, 16);
      Lrow[r] = Lrow[r] * alpha + rs;
      Mrow[r] = Mn;
      for (int t = 0; t < 4; ++t) O[t][r] *= alpha;
    }

    // drain P stores (dscnt) and V async copies (asynccnt) before fragment gathers
    wait_dscnt0();
    wait_asynccnt0();

    ABFrag pf;
    pf.q[0] = *(const uint4*)(&sP[mr * 40 + kb]);
    pf.q[1] = *(const uint4*)(&sP[mr * 40 + kb + 16]);

#pragma unroll
    for (int t = 0; t < 4; ++t) {
      ABFrag vf;
#pragma unroll
      for (int e = 0; e < 16; ++e) {
        int d = kb + (e & 7) + ((e >> 3) << 4);
        vf.v[e] = sV[d * 72 + t * 16 + mr];
      }
      O[t] = wmma_bf16(pf.v, vf.v, O[t]);
    }
    wait_dscnt0();   // V gathers done before next chunk's async copies overwrite sV
  }

  for (int t = 0; t < 4; ++t)
    for (int r = 0; r < 8; ++r) {
      int m = r + 8 * half;
      float o = O[t][r] / Lrow[r];
      Ctx[(size_t)(bt0 + m) * (HQ * HD) + hq * HD + t * 16 + mr] = (bf16)o;
    }
}

// ---------------- kurtosis ----------------
__global__ void scalar_init_kernel(float* out, const float* ksum, int idx) {
  if (threadIdx.x == 0 && blockIdx.x == 0) out[idx] = ksum[0];
}

__global__ __launch_bounds__(256) void kurtosis_kernel(const float* __restrict__ X,
                                                       float* __restrict__ out, int sidx) {
  int row = blockIdx.x;
  const float* x = X + (size_t)row * C_DIM;
  float s1 = 0.f;
  for (int c = threadIdx.x; c < C_DIM; c += 256) s1 += x[c];
  float mean = block_reduce_sum_256(s1) * (1.0f / C_DIM);
  float s2 = 0.f, s4 = 0.f;
  for (int c = threadIdx.x; c < C_DIM; c += 256) {
    float d = x[c] - mean;
    float d2 = d * d;
    s2 += d2; s4 += d2 * d2;
  }
  float var = block_reduce_sum_256(s2) * (1.0f / C_DIM);
  float m4  = block_reduce_sum_256(s4) * (1.0f / C_DIM);
  if (threadIdx.x == 0) {
    float kurt = m4 / (var * var + 1e-6f) - 3.0f;
    if (kurt > 0.f) atomicAdd(&out[sidx], kurt);
  }
}

// ---------------- host launch ----------------
extern "C" void kernel_launch(void* const* d_in, const int* in_sizes, int n_in,
                              void* d_out, int out_size, void* d_ws, size_t ws_size,
                              hipStream_t stream) {
  const float* x     = (const float*)d_in[0];
  const float* ksum  = (const float*)d_in[1];
  const float* rms1  = (const float*)d_in[2];
  const float* wq_f  = (const float*)d_in[3];
  const float* wk_f  = (const float*)d_in[4];
  const float* wv_f  = (const float*)d_in[5];
  const float* wo_f  = (const float*)d_in[6];
  const float* rms2  = (const float*)d_in[7];
  const float* wg_f  = (const float*)d_in[8];
  const float* wu_f  = (const float*)d_in[9];
  const float* wd_f  = (const float*)d_in[10];

  char* ws = (char*)d_ws;
  size_t off = 0;
  auto alloc = [&](size_t bytes) -> char* {
    char* p = ws + off;
    off = (off + bytes + 255) & ~(size_t)255;
    return p;
  };

  bf16*  attn_in = (bf16*)alloc((size_t)NT * C_DIM * 2);
  float* qraw    = (float*)alloc((size_t)NT * 1024 * 4);
  float* kraw    = (float*)alloc((size_t)NT * 512 * 4);
  bf16*  qrope   = (bf16*)alloc((size_t)NT * 1024 * 2);
  bf16*  krope   = (bf16*)alloc((size_t)NT * 512 * 2);
  bf16*  vbf     = (bf16*)alloc((size_t)NT * 512 * 2);
  bf16*  ctx     = (bf16*)alloc((size_t)NT * 1024 * 2);
  float* x1      = (float*)alloc((size_t)NT * C_DIM * 4);
  bf16*  mlp_in  = (bf16*)alloc((size_t)NT * C_DIM * 2);
  bf16*  hbuf    = (bf16*)alloc((size_t)NT * HIDDEN * 2);
  bf16*  wq = (bf16*)alloc((size_t)1024 * 1024 * 2);
  bf16*  wk = (bf16*)alloc((size_t)1024 * 512 * 2);
  bf16*  wv = (bf16*)alloc((size_t)1024 * 512 * 2);
  bf16*  wo = (bf16*)alloc((size_t)1024 * 1024 * 2);
  bf16*  wg = (bf16*)alloc((size_t)1024 * 4096 * 2);
  bf16*  wu = (bf16*)alloc((size_t)1024 * 4096 * 2);
  bf16*  wd = (bf16*)alloc((size_t)4096 * 1024 * 2);

  auto cvt = [&](const float* src, bf16* dst, int n) {
    cvt_bf16_kernel<<<(n + 255) / 256, 256, 0, stream>>>(src, dst, n);
  };
  cvt(wq_f, wq, 1024 * 1024);
  cvt(wk_f, wk, 1024 * 512);
  cvt(wv_f, wv, 1024 * 512);
  cvt(wo_f, wo, 1024 * 1024);
  cvt(wg_f, wg, 1024 * 4096);
  cvt(wu_f, wu, 1024 * 4096);
  cvt(wd_f, wd, 4096 * 1024);

  rmsnorm_kernel<<<NT, 256, 0, stream>>>(x, rms1, attn_in);

  // QKV projections
  gemm_bf16_kernel<NT, 1024, 1024><<<dim3(1024 / 64, NT / 64), 128, 0, stream>>>(attn_in, wq, qraw, nullptr, nullptr);
  gemm_bf16_kernel<NT, 512, 1024><<<dim3(512 / 64, NT / 64), 128, 0, stream>>>(attn_in, wk, kraw, nullptr, nullptr);
  gemm_bf16_kernel<NT, 512, 1024><<<dim3(512 / 64, NT / 64), 128, 0, stream>>>(attn_in, wv, nullptr, vbf, nullptr);

  rope_kernel<<<(NT * 16 * 32 + 255) / 256, 256, 0, stream>>>(qraw, qrope, 16);
  rope_kernel<<<(NT * 8 * 32 + 255) / 256, 256, 0, stream>>>(kraw, krope, 8);

  attn_kernel<<<dim3(T_DIM / 16, HQ, B_DIM), 32, 0, stream>>>(qrope, krope, vbf, ctx);

  // out projection + residual
  gemm_bf16_kernel<NT, 1024, 1024><<<dim3(1024 / 64, NT / 64), 128, 0, stream>>>(ctx, wo, x1, nullptr, x);

  rmsnorm_kernel<<<NT, 256, 0, stream>>>(x1, rms2, mlp_in);

  mlp_gateup_kernel<NT, HIDDEN, 1024><<<dim3(HIDDEN / 64, NT / 64), 128, 0, stream>>>(mlp_in, wg, wu, hbuf);

  // down projection + residual -> d_out
  gemm_bf16_kernel<NT, 1024, HIDDEN><<<dim3(1024 / 64, NT / 64), 128, 0, stream>>>(hbuf, wd, (float*)d_out, nullptr, x1);

  scalar_init_kernel<<<1, 32, 0, stream>>>((float*)d_out, ksum, NT * C_DIM);
  kurtosis_kernel<<<NT, 256, 0, stream>>>((const float*)d_out, (float*)d_out, NT * C_DIM);
}
